// PatchTransformerMSG_15857019257412
// MI455X (gfx1250) — matrix operand order repaired and checked by reading, not compiled, not run on hardware
//
#include <hip/hip_runtime.h>
#include <hip/hip_bf16.h>

typedef __attribute__((ext_vector_type(16))) _Float16 v16h;
typedef __attribute__((ext_vector_type(8)))  float    v8f;
typedef __attribute__((ext_vector_type(4)))  unsigned int v4u;

namespace {

constexpr int   kTok   = 16 * 4096;   // B * S
constexpr int   kDim   = 384;         // TRANS_DIM == PATCH_DIM
constexpr int   kPosH  = 128;
constexpr int   kMTile = 64;          // tokens per workgroup
constexpr float kScale = 0.051031036307982884f; // 384^-0.5
constexpr int   kWMat  = kDim * kDim;

// f16 workspace layout (element offsets), all matrices stored N-major (out-dim major)
constexpr size_t OFF_WIN  = 0;
constexpr size_t OFF_WQ   = 1ull * kWMat;
constexpr size_t OFF_WK   = 2ull * kWMat;
constexpr size_t OFF_WV   = 3ull * kWMat;
constexpr size_t OFF_WG1  = 4ull * kWMat;
constexpr size_t OFF_WG2  = 5ull * kWMat;
constexpr size_t OFF_WOUT = 6ull * kWMat;
constexpr size_t OFF_POS2 = 7ull * kWMat;          // 384 x 128

union FragU { v16h h; v4u u[2]; };

__device__ __forceinline__ float gelu_exact(float a) {
  return 0.5f * a * (1.0f + erff(a * 0.7071067811865475f));
}

// A fragment (16x32 f16, M x K) from row-major f16 LDS tile.
// ISA 7.12.2: lanes 0-15 -> M=lane, K = {k0..k0+7, k0+16..k0+23}
//             lanes 16-31 -> M=lane-16, K = {k0+8..k0+15, k0+24..k0+31}
__device__ __forceinline__ v16h load_frag_A(const _Float16* sA, int m0, int k0, int lda) {
  const int lane = threadIdx.x & 31;
  const int row  = m0 + (lane & 15);
  const int kb   = (lane < 16) ? 0 : 8;
  const _Float16* p = sA + row * lda + k0 + kb;
  FragU f;
  f.u[0] = *reinterpret_cast<const v4u*>(p);        // K = k0+kb .. +7
  f.u[1] = *reinterpret_cast<const v4u*>(p + 16);   // K = k0+16+kb .. +7
  return f.h;
}

// B fragment (32x16 f16, K x N) from N-major f16 weights (Wt[n*ldb + k]).
// lanes 0-15: col = n0+lane, K = {k0..k0+7, k0+16..k0+23} ; lanes 16-31: col = n0+lane-16, K = {k0+8..k0+15, k0+24..k0+31}
__device__ __forceinline__ v16h load_frag_B(const _Float16* Wt, int n0, int k0, int ldb) {
  const int lane = threadIdx.x & 31;
  const int col  = n0 + (lane & 15);
  const int kb   = (lane < 16) ? 0 : 8;
  const _Float16* p = Wt + (size_t)col * ldb + k0 + kb;
  FragU f;
  f.u[0] = *reinterpret_cast<const v4u*>(p);        // K = k0+kb .. +7
  f.u[1] = *reinterpret_cast<const v4u*>(p + 16);   // K = k0+16+kb .. +7
  return f.h;
}

__device__ __forceinline__ v8f wmma_f16(v16h a, v16h b, v8f c) {
  return __builtin_amdgcn_wmma_f32_16x16x32_f16(false, a, false, b, (short)0, c,
                                                false, false);
}

// One wave computes 12 output 16x16 tiles of a (64 x 384) = (64 x K) @ (K x 384) GEMM.
// Row tile = wid&3, column tiles = (wid>>2)*12 .. +11. A fragments cached in registers.
template <int KT, typename Epi>
__device__ __forceinline__ void wave_gemm(const _Float16* __restrict__ sA, int lda,
                                          const _Float16* __restrict__ Wt, int ldb,
                                          Epi&& epi) {
  const int wid = threadIdx.x >> 5;
  const int m0  = (wid & 3) * 16;
  const int c0  = (wid >> 2) * 12;
  v16h aF[KT];
#pragma unroll
  for (int t = 0; t < KT; ++t) aF[t] = load_frag_A(sA, m0, t * 32, lda);
  for (int j = 0; j < 12; ++j) {
    const int n0 = (c0 + j) * 16;
    if (j + 1 < 12)   // pull next column panel of weights toward L2/L0
      __builtin_prefetch(Wt + (size_t)(n0 + 16) * ldb, 0, 1);
    v8f acc = {};
#pragma unroll
    for (int t = 0; t < KT; ++t)
      acc = wmma_f16(aF[t], load_frag_B(Wt, n0, t * 32, ldb), acc);
    epi(acc, m0, n0);
  }
}

__global__ void transpose_f32_to_f16(const float* __restrict__ src,
                                     _Float16* __restrict__ dst, int K, int N) {
  const int idx = blockIdx.x * blockDim.x + threadIdx.x;
  if (idx < K * N) {
    const int n = idx / K;
    const int k = idx - n * K;
    dst[idx] = (_Float16)src[(size_t)k * N + n];  // dst is N-major
  }
}

__global__ __launch_bounds__(256, 1) void fused_patch_transformer(
    const float* __restrict__ x,            // (kTok, 3 + 384)
    const _Float16* __restrict__ ws,        // transposed f16 weights
    const float* __restrict__ Wpos1,        // (3, 128) row-major f32
    const float* __restrict__ bpos1, const float* __restrict__ bpos2,
    const float* __restrict__ bin,   const float* __restrict__ bg1,
    const float* __restrict__ bg2,   const float* __restrict__ bout,
    float* __restrict__ qout, float* __restrict__ res) {
  extern __shared__ char smem_raw[];
  _Float16* sPre = reinterpret_cast<_Float16*>(smem_raw);   // 64x384 pre (f16)
  _Float16* sH   = sPre + kMTile * kDim;                    // 64x384 h, then gelu(g1) temp
  _Float16* sQK  = sH   + kMTile * kDim;                    // pos_emb -> (q-k) -> gate -> attn*v
  _Float16* sV   = sQK  + kMTile * kDim;                    // 64x384 v
  _Float16* sPH  = sV   + kMTile * kDim;                    // 64x128 gelu(pos layer1)
  float* sRedMax = reinterpret_cast<float*>(sPH + kMTile * kPosH); // 64x4
  float* sRedSum = sRedMax + kMTile * 4;

  const int tokBase = blockIdx.x * kMTile;

  // ---- Stage 0: stage pre (f16) and the tiny 3->128 pos MLP layer (VALU) ----
  for (int i = threadIdx.x; i < kMTile * kDim; i += 256) {
    const int r = i / kDim, c = i - (i / kDim) * kDim;
    sPre[i] = (_Float16)x[(size_t)(tokBase + r) * 387 + 3 + c];
  }
  for (int i = threadIdx.x; i < kMTile * kPosH; i += 256) {
    const int r = i / kPosH, c = i - (i / kPosH) * kPosH;
    const float* cx = x + (size_t)(tokBase + r) * 387;
    const float a = cx[0] * Wpos1[c] + cx[1] * Wpos1[kPosH + c] +
                    cx[2] * Wpos1[2 * kPosH + c] + bpos1[c];
    sPH[i] = (_Float16)gelu_exact(a);
  }
  __syncthreads();

  // ---- Stage 1: pos_emb = posh @ W_pos2 + b_pos2  -> sQK (f16) ----
  wave_gemm<4>(sPH, kPosH, ws + OFF_POS2, kPosH, [&](v8f acc, int m0, int n0) {
    const int lane = threadIdx.x & 31;
    const int col  = n0 + (lane & 15);
    const int rb   = (lane < 16) ? 0 : 8;
    const float bb = bpos2[col];
#pragma unroll
    for (int v = 0; v < 8; ++v)
      sQK[(m0 + rb + v) * kDim + col] = (_Float16)(acc[v] + bb);
  });
  __syncthreads();

  // ---- Stage 2: h = pre @ W_in + b_in + pos_emb -> sH (f16) ----
  wave_gemm<12>(sPre, kDim, ws + OFF_WIN, kDim, [&](v8f acc, int m0, int n0) {
    const int lane = threadIdx.x & 31;
    const int col  = n0 + (lane & 15);
    const int rb   = (lane < 16) ? 0 : 8;
    const float bb = bin[col];
#pragma unroll
    for (int v = 0; v < 8; ++v) {
      const int r = m0 + rb + v;
      sH[r * kDim + col] = (_Float16)(acc[v] + bb + (float)sQK[r * kDim + col]);
    }
  });
  __syncthreads();

  // ---- Stage 3: q,k,v fused (shared A fragments). q_out -> global, q-k -> sQK, v -> sV ----
  {
    const int wid = threadIdx.x >> 5;
    const int m0  = (wid & 3) * 16;
    const int c0  = (wid >> 2) * 12;
    v16h aF[12];
#pragma unroll
    for (int t = 0; t < 12; ++t) aF[t] = load_frag_A(sH, m0, t * 32, kDim);
    for (int j = 0; j < 12; ++j) {
      const int n0 = (c0 + j) * 16;
      v8f aq = {}, ak = {}, av = {};
#pragma unroll
      for (int t = 0; t < 12; ++t) {
        const int k0 = t * 32;
        aq = wmma_f16(aF[t], load_frag_B(ws + OFF_WQ, n0, k0, kDim), aq);
        ak = wmma_f16(aF[t], load_frag_B(ws + OFF_WK, n0, k0, kDim), ak);
        av = wmma_f16(aF[t], load_frag_B(ws + OFF_WV, n0, k0, kDim), av);
      }
      const int lane = threadIdx.x & 31;
      const int col  = n0 + (lane & 15);
      const int rb   = (lane < 16) ? 0 : 8;
#pragma unroll
      for (int v = 0; v < 8; ++v) {
        const int r = m0 + rb + v;
        qout[(size_t)(tokBase + r) * kDim + col] = aq[v];
        sQK[r * kDim + col] = (_Float16)(aq[v] - ak[v]);
        sV[r * kDim + col]  = (_Float16)av[v];
      }
    }
  }
  __syncthreads();

  // ---- Stage 4: t = gelu((q-k) @ W_g1 + b_g1) -> sH (f16) ----
  wave_gemm<12>(sQK, kDim, ws + OFF_WG1, kDim, [&](v8f acc, int m0, int n0) {
    const int lane = threadIdx.x & 31;
    const int col  = n0 + (lane & 15);
    const int rb   = (lane < 16) ? 0 : 8;
    const float bb = bg1[col];
#pragma unroll
    for (int v = 0; v < 8; ++v)
      sH[(m0 + rb + v) * kDim + col] = (_Float16)gelu_exact(acc[v] + bb);
  });
  __syncthreads();

  // ---- Stage 5: gate = t @ W_g2 + b_g2 -> sQK (f16) ----
  wave_gemm<12>(sH, kDim, ws + OFF_WG2, kDim, [&](v8f acc, int m0, int n0) {
    const int lane = threadIdx.x & 31;
    const int col  = n0 + (lane & 15);
    const int rb   = (lane < 16) ? 0 : 8;
    const float bb = bg2[col];
#pragma unroll
    for (int v = 0; v < 8; ++v)
      sQK[(m0 + rb + v) * kDim + col] = (_Float16)(acc[v] + bb);
  });
  __syncthreads();

  // ---- Stage 6: softmax over feature dim (384) fused with attn*v, in place in sQK ----
  {
    const int row  = threadIdx.x >> 2;          // 4 threads per row
    const int part = threadIdx.x & 3;
    const _Float16* g = sQK + row * kDim + part * 96;
    float mx = -3.0e38f;
    for (int i = 0; i < 96; ++i) mx = fmaxf(mx, (float)g[i] * kScale);
    sRedMax[row * 4 + part] = mx;
    __syncthreads();
    mx = fmaxf(fmaxf(sRedMax[row * 4 + 0], sRedMax[row * 4 + 1]),
               fmaxf(sRedMax[row * 4 + 2], sRedMax[row * 4 + 3]));
    float sum = 0.0f;
    for (int i = 0; i < 96; ++i) sum += expf((float)g[i] * kScale - mx);
    sRedSum[row * 4 + part] = sum;
    __syncthreads();
    sum = sRedSum[row * 4 + 0] + sRedSum[row * 4 + 1] +
          sRedSum[row * 4 + 2] + sRedSum[row * 4 + 3];
    const float inv = 1.0f / sum;
    _Float16* gw = sQK + row * kDim + part * 96;
    const _Float16* vv = sV + row * kDim + part * 96;
    for (int i = 0; i < 96; ++i) {
      const float e = expf((float)gw[i] * kScale - mx) * inv;
      gw[i] = (_Float16)(e * (float)vv[i]);
    }
  }
  __syncthreads();

  // ---- Stage 7: res = (attn*v) @ W_out + b_out + pre -> global ----
  wave_gemm<12>(sQK, kDim, ws + OFF_WOUT, kDim, [&](v8f acc, int m0, int n0) {
    const int lane = threadIdx.x & 31;
    const int col  = n0 + (lane & 15);
    const int rb   = (lane < 16) ? 0 : 8;
    const float bb = bout[col];
#pragma unroll
    for (int v = 0; v < 8; ++v) {
      const int r = m0 + rb + v;
      res[(size_t)(tokBase + r) * kDim + col] =
          acc[v] + bb + (float)sPre[r * kDim + col];
    }
  });
}

}  // namespace

extern "C" void kernel_launch(void* const* d_in, const int* in_sizes, int n_in,
                              void* d_out, int out_size, void* d_ws, size_t ws_size,
                              hipStream_t stream) {
  // setup_inputs order:
  // 0 idx, 1 q(unused), 2 x, 3 W_pos1, 4 b_pos1, 5 W_pos2, 6 b_pos2, 7 W_in, 8 b_in,
  // 9 W_q, 10 W_k, 11 W_v, 12 W_g1, 13 b_g1, 14 W_g2, 15 b_g2, 16 W_out, 17 b_out
  const float* x     = (const float*)d_in[2];
  const float* Wpos1 = (const float*)d_in[3];
  const float* bpos1 = (const float*)d_in[4];
  const float* Wpos2 = (const float*)d_in[5];
  const float* bpos2 = (const float*)d_in[6];
  const float* Win   = (const float*)d_in[7];
  const float* bin   = (const float*)d_in[8];
  const float* Wq    = (const float*)d_in[9];
  const float* Wk    = (const float*)d_in[10];
  const float* Wv    = (const float*)d_in[11];
  const float* Wg1   = (const float*)d_in[12];
  const float* bg1   = (const float*)d_in[13];
  const float* Wg2   = (const float*)d_in[14];
  const float* bg2   = (const float*)d_in[15];
  const float* Wout  = (const float*)d_in[16];
  const float* bout  = (const float*)d_in[17];

  _Float16* ws = (_Float16*)d_ws;

  // Prep: f32 -> f16 + transpose to N-major (weights stay L2-resident: ~2.2 MB)
  const int tpb = 256;
  const int nW  = (kWMat + tpb - 1) / tpb;
  transpose_f32_to_f16<<<nW, tpb, 0, stream>>>(Win,  ws + OFF_WIN,  kDim, kDim);
  transpose_f32_to_f16<<<nW, tpb, 0, stream>>>(Wq,   ws + OFF_WQ,   kDim, kDim);
  transpose_f32_to_f16<<<nW, tpb, 0, stream>>>(Wk,   ws + OFF_WK,   kDim, kDim);
  transpose_f32_to_f16<<<nW, tpb, 0, stream>>>(Wv,   ws + OFF_WV,   kDim, kDim);
  transpose_f32_to_f16<<<nW, tpb, 0, stream>>>(Wg1,  ws + OFF_WG1,  kDim, kDim);
  transpose_f32_to_f16<<<nW, tpb, 0, stream>>>(Wg2,  ws + OFF_WG2,  kDim, kDim);
  transpose_f32_to_f16<<<nW, tpb, 0, stream>>>(Wout, ws + OFF_WOUT, kDim, kDim);
  const int nP = (kDim * kPosH + tpb - 1) / tpb;
  transpose_f32_to_f16<<<nP, tpb, 0, stream>>>(Wpos2, ws + OFF_POS2, kPosH, kDim);

  float* qout = (float*)d_out;
  float* res  = qout + (size_t)kTok * kDim;

  // Dynamic LDS: 4 f16 buffers (64x384) + 1 f16 buffer (64x128) + 2 f32 reduce arrays
  const size_t smem = (size_t)(4 * kMTile * kDim + kMTile * kPosH) * sizeof(_Float16)
                    + (size_t)kMTile * 8 * sizeof(float);

  fused_patch_transformer<<<kTok / kMTile, 256, smem, stream>>>(
      x, ws, Wpos1, bpos1, bpos2, bin, bg1, bg2, bout, qout, res);
}
